// GNN_HSG_91242285236266
// MI455X (gfx1250) — compile-verified, matrix operand order checked by reading, not run on hardware
//
#include <hip/hip_runtime.h>
#include <hip/hip_bf16.h>

typedef __attribute__((ext_vector_type(16))) _Float16 v16h;
typedef __attribute__((ext_vector_type(8)))  _Float16 v8h;
typedef __attribute__((ext_vector_type(8)))  float    v8f;

#define N_NODES 12288
#define N_FEAT  512
#define NHID    256

// ---------------------------------------------------------------------------
// WMMA fragment loader (16-bit, 16x32 tile) from a row-major f16 matrix.
// ISA layout (CDNA5 ISA): lane L holds row (L&15); K elements
// [kb, kb+8) in v16h elems 0-7 and [kb+16, kb+24) in elems 8-15, kb=(L>>4)*8.
// B (32x16) uses the same per-lane pattern with "row" -> "column", so passing
// the transposed-B (row-major N x K) here yields the correct B fragment.
// Works for global or LDS pointers (addrspace inferred after inlining).
// ---------------------------------------------------------------------------
__device__ __forceinline__ v16h load_frag(const _Float16* __restrict__ p,
                                          int ld, int row0, int k0) {
  int lane = threadIdx.x & 31;
  const _Float16* base =
      p + (size_t)(row0 + (lane & 15)) * ld + (k0 + ((lane >> 4) << 3));
  v8h lo = *(const v8h*)(base);
  v8h hi = *(const v8h*)(base + 16);
  v16h out;
#pragma unroll
  for (int i = 0; i < 8; ++i) { out[i] = lo[i]; out[i + 8] = hi[i]; }
  return out;
}

// ---------------------------------------------------------------------------
// CDNA5 async global->LDS staging (ASYNCcnt path, CDNA5 ISA)
// ---------------------------------------------------------------------------
__device__ __forceinline__ void async_b128(unsigned lds_addr, unsigned long long gaddr) {
  asm volatile("global_load_async_to_lds_b128 %0, %1, off"
               :: "v"(lds_addr), "v"(gaddr) : "memory");
}
__device__ __forceinline__ void wait_async4() {
#if __has_builtin(__builtin_amdgcn_s_wait_asynccnt)
  __builtin_amdgcn_s_wait_asynccnt(4);
#else
  asm volatile("s_wait_asynccnt 0x4" ::: "memory");
#endif
}
__device__ __forceinline__ void wait_async0() {
#if __has_builtin(__builtin_amdgcn_s_wait_asynccnt)
  __builtin_amdgcn_s_wait_asynccnt(0);
#else
  asm volatile("s_wait_asynccnt 0x0" ::: "memory");
#endif
}

// ---------------------------------------------------------------------------
// Small utility kernels
// ---------------------------------------------------------------------------
__global__ void fill_f32(float* p, size_t n) {
  for (size_t i = (size_t)blockIdx.x * blockDim.x + threadIdx.x; i < n;
       i += (size_t)gridDim.x * blockDim.x) p[i] = 0.0f;
}
__global__ void fill_f16(_Float16* p, size_t n) {
  for (size_t i = (size_t)blockIdx.x * blockDim.x + threadIdx.x; i < n;
       i += (size_t)gridDim.x * blockDim.x) p[i] = (_Float16)0.0f;
}
// wT[n][k] = (f16) w[k][n]   (prepares transposed-B for WMMA)
__global__ void transpose_to_h(const float* __restrict__ w, _Float16* __restrict__ wT,
                               int K, int N) {
  size_t total = (size_t)K * N;
  for (size_t i = (size_t)blockIdx.x * blockDim.x + threadIdx.x; i < total;
       i += (size_t)gridDim.x * blockDim.x) {
    int k = (int)(i / N), n = (int)(i % N);
    wT[(size_t)n * K + k] = (_Float16)w[i];
  }
}
__global__ void convert_to_h(const float* __restrict__ s, _Float16* __restrict__ d, size_t n) {
  for (size_t i = (size_t)blockIdx.x * blockDim.x + threadIdx.x; i < n;
       i += (size_t)gridDim.x * blockDim.x) d[i] = (_Float16)s[i];
}
__global__ void add_vec(const float* __restrict__ a, const float* __restrict__ b,
                        float* __restrict__ d, int n) {
  int i = blockIdx.x * blockDim.x + threadIdx.x;
  if (i < n) d[i] = a[i] + b[i];
}

// ---------------------------------------------------------------------------
// Row L2-normalize x [N, 512] -> f16
// ---------------------------------------------------------------------------
__global__ void normalize_kernel(const float* __restrict__ x, _Float16* __restrict__ xh) {
  __shared__ float red[256];
  int row = blockIdx.x;
  const float* xr = x + (size_t)row * N_FEAT;
  float s = 0.0f;
  for (int c = threadIdx.x; c < N_FEAT; c += 256) { float v = xr[c]; s += v * v; }
  red[threadIdx.x] = s;
  __syncthreads();
  for (int off = 128; off > 0; off >>= 1) {
    if (threadIdx.x < off) red[threadIdx.x] += red[threadIdx.x + off];
    __syncthreads();
  }
  float inv = 1.0f / sqrtf(red[0]);
  for (int c = threadIdx.x; c < N_FEAT; c += 256)
    xh[(size_t)row * N_FEAT + c] = (_Float16)(xr[c] * inv);
}

// ---------------------------------------------------------------------------
// Cosine-similarity argmin ("farthest neighbor") via f16 WMMA.
// Grid: (96 row-blocks) x (4 j-quarters). Block = 8 waves; wave w owns the
// 16-row A tile rowblk+16w (16 fragments, register-resident). All waves walk
// the same column tiles; each 16x512 B slab is staged into LDS with async
// b128 copies (ASYNCcnt), double-buffered so the next slab's DMA overlaps the
// current slab's 16 WMMAs per wave. LDS B rows padded to 520 halves
// (row stride 1040B = 260 dwords = 4 mod 64 banks -> conflict-free b128).
// Per-quarter argmin partials are combined by far_reduce_kernel.
// ---------------------------------------------------------------------------
#define LDB 520
__global__ void __launch_bounds__(256)
sim_argmin_kernel(const _Float16* __restrict__ xh, float* __restrict__ pv,
                  int* __restrict__ pi) {
  const int K = N_FEAT;
  __shared__ __align__(16) _Float16 btile[2 * 16 * LDB];   // ~33 KB

  int wave = threadIdx.x >> 5;
  int lane = threadIdx.x & 31;
  int rowblk = blockIdx.x * 128;
  int jq = blockIdx.y;
  const int jbeg = jq * (N_NODES / 16 / 4);
  const int jend = jbeg + (N_NODES / 16 / 4);

  // A tile for this wave: 16 rows, kept in 16 register fragments (~128 VGPRs)
  int row0 = rowblk + wave * 16;
  v16h afrag[16];
#pragma unroll
  for (int kk = 0; kk < 16; ++kk) afrag[kk] = load_frag(xh, K, row0, kk * 32);

  const unsigned bbase = (unsigned)(uintptr_t)(void*)btile;
  const unsigned long long gbase = (unsigned long long)(uintptr_t)xh;

  // issue async copy of one 16x512 B slab (16KB): 4 x b128 per thread
  auto issue_tile = [&](int j, int buf) {
    unsigned long long g = gbase + (unsigned long long)j * (16ull * N_FEAT * 2ull);
    unsigned lb = bbase + (unsigned)buf * (16u * LDB * 2u);
#pragma unroll
    for (int i = 0; i < 4; ++i) {
      unsigned c = threadIdx.x + 256u * i;            // chunk id: row = c>>6
      unsigned lds = lb + c * 16u + (c >> 6) * 16u;   // +16B pad per row
      async_b128(lds, g + (unsigned long long)c * 16u);
    }
  };

  float minv[8]; int mini[8];
#pragma unroll
  for (int r = 0; r < 8; ++r) { minv[r] = 3.4e38f; mini[r] = 0; }

  issue_tile(jbeg, 0);
  for (int j = jbeg; j < jend; ++j) {
    int buf = (j - jbeg) & 1;
    if (j + 1 < jend) { issue_tile(j + 1, buf ^ 1); wait_async4(); }
    else              { wait_async0(); }
    __syncthreads();                       // B slab j fully in LDS for all waves

    const _Float16* bt = btile + buf * (16 * LDB);
    v8f acc = {};
#pragma unroll
    for (int kk = 0; kk < 16; ++kk) {
      v16h b = load_frag(bt, LDB, 0, kk * 32);
      acc = __builtin_amdgcn_wmma_f32_16x16x32_f16(false, afrag[kk], false, b,
                                                   (short)0, acc, false, false);
    }
    int col = (j << 4) + (lane & 15);
#pragma unroll
    for (int r = 0; r < 8; ++r) {
      float v = acc[r];
      if (v < minv[r] || (v == minv[r] && col < mini[r])) { minv[r] = v; mini[r] = col; }
    }
    __syncthreads();                       // all waves done reading before overwrite
  }

  // cross-lane argmin: masks 8,4,2,1 stay within each 16-lane half, which
  // holds one row's 16 column candidates (C layout: N across lanes).
#pragma unroll
  for (int r = 0; r < 8; ++r) {
    float v = minv[r]; int ix = mini[r];
#pragma unroll
    for (int m = 8; m >= 1; m >>= 1) {
      float ov = __shfl_xor(v, m, 32);
      int   oi = __shfl_xor(ix, m, 32);
      if (ov < v || (ov == v && oi < ix)) { v = ov; ix = oi; }
    }
    if ((lane & 15) == 0) {
      int row = row0 + ((lane >> 4) << 3) + r;
      pv[row * 4 + jq] = v; pi[row * 4 + jq] = ix;
    }
  }
}

// combine the 4 j-quarter partials (first-occurrence tie-break like argmin)
__global__ void far_reduce_kernel(const float* __restrict__ pv, const int* __restrict__ pi,
                                  int* __restrict__ far) {
  int row = blockIdx.x * blockDim.x + threadIdx.x;
  if (row >= N_NODES) return;
  float bv = pv[row * 4]; int bi = pi[row * 4];
#pragma unroll
  for (int q = 1; q < 4; ++q) {
    float v = pv[row * 4 + q]; int ix = pi[row * 4 + q];
    if (v < bv || (v == bv && ix < bi)) { bv = v; bi = ix; }
  }
  far[row] = bi;
}

// ---------------------------------------------------------------------------
// GIN aggregation helpers
// ---------------------------------------------------------------------------
__global__ void colsum_kernel(const float* __restrict__ h, float* __restrict__ colsum, int C) {
  int r0 = blockIdx.x * 64;
  for (int c = threadIdx.x; c < C; c += blockDim.x) {
    float s = 0.0f;
    for (int r = 0; r < 64; ++r) s += h[(size_t)(r0 + r) * C + c];
    atomicAdd(&colsum[c], s);
  }
}
__global__ void scatter_kernel(const float* __restrict__ h, const int* __restrict__ far,
                               float* __restrict__ excl, int C) {
  size_t total = (size_t)N_NODES * C;
  for (size_t i = (size_t)blockIdx.x * blockDim.x + threadIdx.x; i < total;
       i += (size_t)gridDim.x * blockDim.x) {
    int row = (int)(i / C), col = (int)(i % C);
    atomicAdd(&excl[(size_t)far[row] * C + col], h[i]);
  }
}
// z = h + colsum - excl  -> f16
__global__ void agg_kernel(const float* __restrict__ h, const float* __restrict__ colsum,
                           const float* __restrict__ excl, _Float16* __restrict__ zh, int C) {
  size_t total = (size_t)N_NODES * C;
  for (size_t i = (size_t)blockIdx.x * blockDim.x + threadIdx.x; i < total;
       i += (size_t)gridDim.x * blockDim.x) {
    int col = (int)(i % C);
    zh[i] = (_Float16)(h[i] + colsum[col] - excl[i]);
  }
}

// ---------------------------------------------------------------------------
// Generic f16 WMMA GEMM: C[M,N] = act( A[M,K] @ B[K,N] + bias_scale*bias )
// BT is B transposed, row-major [N,K]. DUAL: += A2 @ B2 (for LSTM).
// One 16x16 output tile per wave, 8 waves / block; consecutive waves share
// the same A row tile (WGP$-broadcast), weights stay L2-resident.
// ---------------------------------------------------------------------------
template <bool DUAL, bool RELU>
__global__ void __launch_bounds__(256)
gemm_kernel(const _Float16* __restrict__ A, const _Float16* __restrict__ BT,
            const _Float16* __restrict__ A2, const _Float16* __restrict__ BT2,
            const float* __restrict__ bias, float bias_scale,
            float* __restrict__ Cf, _Float16* __restrict__ Ch,
            int M, int N, int K) {
  int wave = threadIdx.x >> 5;
  int lane = threadIdx.x & 31;
  int tile = blockIdx.x * 8 + wave;
  int tn_count = N >> 4;
  int tm = tile / tn_count, tn = tile % tn_count;
  if (tm >= (M >> 4)) return;  // wave-uniform: EXEC stays all-ones inside WMMA
  int m0 = tm << 4, n0 = tn << 4;

  v8f acc = {};
  for (int k0 = 0; k0 < K; k0 += 32) {
    v16h a = load_frag(A, K, m0, k0);
    v16h b = load_frag(BT, K, n0, k0);
    acc = __builtin_amdgcn_wmma_f32_16x16x32_f16(false, a, false, b, (short)0, acc, false, false);
    if (DUAL) {
      v16h a2 = load_frag(A2, K, m0, k0);
      v16h b2 = load_frag(BT2, K, n0, k0);
      acc = __builtin_amdgcn_wmma_f32_16x16x32_f16(false, a2, false, b2, (short)0, acc, false, false);
    }
  }

  int col = n0 + (lane & 15);
  int rbase = m0 + ((lane >> 4) << 3);
  float bv = bias_scale * bias[col];
#pragma unroll
  for (int r = 0; r < 8; ++r) {
    float v = acc[r] + bv;
    if (RELU) v = fmaxf(v, 0.0f);
    size_t off = (size_t)(rbase + r) * N + col;
    if (Cf) Cf[off] = v;
    if (Ch) Ch[off] = (_Float16)v;
  }
}

// ---------------------------------------------------------------------------
// LSTM gates (PyTorch order i,f,g,o). G:[N,1024] pre-activation.
// c updated in place (f32); h written f16; optional running sum for mean.
// ---------------------------------------------------------------------------
__device__ __forceinline__ float sigmoidf(float x) { return 1.0f / (1.0f + expf(-x)); }

__global__ void lstm_gate_kernel(const float* __restrict__ G, float* __restrict__ c,
                                 _Float16* __restrict__ hh, float* __restrict__ hsum) {
  size_t total = (size_t)N_NODES * NHID;
  for (size_t idx = (size_t)blockIdx.x * blockDim.x + threadIdx.x; idx < total;
       idx += (size_t)gridDim.x * blockDim.x) {
    int row = (int)(idx / NHID), j = (int)(idx % NHID);
    const float* g = G + (size_t)row * (4 * NHID);
    float gi = sigmoidf(g[j]);
    float gf = sigmoidf(g[j + NHID]);
    float gg = tanhf(g[j + 2 * NHID]);
    float go = sigmoidf(g[j + 3 * NHID]);
    float cn = gf * c[idx] + gi * gg;
    c[idx] = cn;
    float hn = go * tanhf(cn);
    hh[idx] = (_Float16)hn;
    if (hsum) hsum[idx] += hn;
  }
}

// ---------------------------------------------------------------------------
// Residuals + stack stats, fused:
//   x2'=x2+x1; x3'=x3+x2'; x4'=x4+x3'; sum=x1+x2'+x3'+x4'
//   s0..s4 = f16 LSTM inputs; partial = avg + max = 0.4*sum + max(sum,x1..x4')
// ---------------------------------------------------------------------------
__global__ void stats_kernel(const float* __restrict__ x1, const float* __restrict__ x2,
                             const float* __restrict__ x3, const float* __restrict__ x4,
                             _Float16* __restrict__ s0, _Float16* __restrict__ s1,
                             _Float16* __restrict__ s2, _Float16* __restrict__ s3,
                             _Float16* __restrict__ s4, float* __restrict__ partial) {
  size_t total = (size_t)N_NODES * NHID;
  for (size_t i = (size_t)blockIdx.x * blockDim.x + threadIdx.x; i < total;
       i += (size_t)gridDim.x * blockDim.x) {
    float a = x1[i];
    float b = x2[i] + a;
    float c = x3[i] + b;
    float d = x4[i] + c;
    float sum = a + b + c + d;
    float avg = sum * 0.4f;  // (sum + a+b+c+d)/5
    float mx = fmaxf(sum, fmaxf(fmaxf(a, b), fmaxf(c, d)));
    s0[i] = (_Float16)sum; s1[i] = (_Float16)a; s2[i] = (_Float16)b;
    s3[i] = (_Float16)c;   s4[i] = (_Float16)d;
    partial[i] = avg + mx;
  }
}

// S = f16(partial + lstm_sum/5)
__global__ void final_s_kernel(const float* __restrict__ partial, const float* __restrict__ lsum,
                               _Float16* __restrict__ sh) {
  size_t total = (size_t)N_NODES * NHID;
  for (size_t i = (size_t)blockIdx.x * blockDim.x + threadIdx.x; i < total;
       i += (size_t)gridDim.x * blockDim.x)
    sh[i] = (_Float16)(partial[i] + 0.2f * lsum[i]);
}

// ---------------------------------------------------------------------------
// Host launcher
// ---------------------------------------------------------------------------
static inline int igrid(size_t n, int b) { size_t g = (n + b - 1) / b; return (int)(g > 16384 ? 16384 : g); }

extern "C" void kernel_launch(void* const* d_in, const int* in_sizes, int n_in,
                              void* d_out, int out_size, void* d_ws, size_t ws_size,
                              hipStream_t stream) {
  (void)in_sizes; (void)n_in; (void)out_size; (void)ws_size;
  const float* x = (const float*)d_in[0];
  const float* conv_w1[4] = {(const float*)d_in[1], (const float*)d_in[5], (const float*)d_in[9],  (const float*)d_in[13]};
  const float* conv_b1[4] = {(const float*)d_in[2], (const float*)d_in[6], (const float*)d_in[10], (const float*)d_in[14]};
  const float* conv_w2[4] = {(const float*)d_in[3], (const float*)d_in[7], (const float*)d_in[11], (const float*)d_in[15]};
  const float* conv_b2[4] = {(const float*)d_in[4], (const float*)d_in[8], (const float*)d_in[12], (const float*)d_in[16]};
  const float* mlp_w = (const float*)d_in[17];
  const float* mlp_b = (const float*)d_in[18];
  const float* lstm_wih[2] = {(const float*)d_in[19], (const float*)d_in[23]};
  const float* lstm_whh[2] = {(const float*)d_in[20], (const float*)d_in[24]};
  const float* lstm_bih[2] = {(const float*)d_in[21], (const float*)d_in[25]};
  const float* lstm_bhh[2] = {(const float*)d_in[22], (const float*)d_in[26]};

  const size_t NH = (size_t)N_NODES * NHID;   // 3.1M
  const size_t NF = (size_t)N_NODES * N_FEAT; // 6.3M

  // --- carve workspace (fixed offsets, 256B aligned) ---
  char* w = (char*)d_ws;
  auto carve = [&](size_t bytes) { void* p = (void*)w; w += (bytes + 255) & ~(size_t)255; return p; };
  _Float16* XH   = (_Float16*)carve(NF * 2);
  int*      FAR  = (int*)carve(N_NODES * 4);
  float*    PV   = (float*)carve((size_t)N_NODES * 4 * 4);
  int*      PI   = (int*)carve((size_t)N_NODES * 4 * 4);
  _Float16* W1T[4]; _Float16* W2T[4];
  W1T[0] = (_Float16*)carve((size_t)256 * 512 * 2);
  for (int c = 1; c < 4; ++c) W1T[c] = (_Float16*)carve((size_t)256 * 256 * 2);
  for (int c = 0; c < 4; ++c) W2T[c] = (_Float16*)carve((size_t)256 * 256 * 2);
  _Float16* MLPT = (_Float16*)carve((size_t)256 * 256 * 2);
  _Float16* WIH[2]; _Float16* WHH[2]; float* BSUM[2];
  for (int l = 0; l < 2; ++l) {
    WIH[l]  = (_Float16*)carve((size_t)1024 * 256 * 2);
    WHH[l]  = (_Float16*)carve((size_t)1024 * 256 * 2);
    BSUM[l] = (float*)carve(1024 * 4);
  }
  float*    COLSUM = (float*)carve(512 * 4);
  float*    EXCL   = (float*)carve(NF * 4);
  _Float16* ZH     = (_Float16*)carve(NF * 2);
  _Float16* TH     = (_Float16*)carve(NH * 2);
  float*    X[4];
  for (int c = 0; c < 4; ++c) X[c] = (float*)carve(NH * 4);
  _Float16* S[5];
  for (int t = 0; t < 5; ++t) S[t] = (_Float16*)carve(NH * 2);
  float*    PARTIAL = (float*)carve(NH * 4);
  float*    G       = (float*)carve((size_t)N_NODES * 1024 * 4);
  float*    C0      = (float*)carve(NH * 4);
  float*    C1      = (float*)carve(NH * 4);
  _Float16* H0      = (_Float16*)carve(NH * 2);
  _Float16* H1      = (_Float16*)carve(NH * 2);
  float*    LSUM    = (float*)carve(NH * 4);
  _Float16* SH      = (_Float16*)carve(NH * 2);

  // --- weight prep ---
  transpose_to_h<<<igrid(512 * 256, 256), 256, 0, stream>>>(conv_w1[0], W1T[0], 512, 256);
  for (int c = 1; c < 4; ++c)
    transpose_to_h<<<igrid(256 * 256, 256), 256, 0, stream>>>(conv_w1[c], W1T[c], 256, 256);
  for (int c = 0; c < 4; ++c)
    transpose_to_h<<<igrid(256 * 256, 256), 256, 0, stream>>>(conv_w2[c], W2T[c], 256, 256);
  transpose_to_h<<<igrid(256 * 256, 256), 256, 0, stream>>>(mlp_w, MLPT, 256, 256);
  for (int l = 0; l < 2; ++l) {
    convert_to_h<<<igrid(1024 * 256, 256), 256, 0, stream>>>(lstm_wih[l], WIH[l], (size_t)1024 * 256);
    convert_to_h<<<igrid(1024 * 256, 256), 256, 0, stream>>>(lstm_whh[l], WHH[l], (size_t)1024 * 256);
    add_vec<<<4, 256, 0, stream>>>(lstm_bih[l], lstm_bhh[l], BSUM[l], 1024);
  }

  // --- graph construction: normalize + async-staged WMMA similarity argmin ---
  normalize_kernel<<<N_NODES, 256, 0, stream>>>(x, XH);
  sim_argmin_kernel<<<dim3(N_NODES / 128, 4), 256, 0, stream>>>(XH, PV, PI);
  far_reduce_kernel<<<igrid(N_NODES, 256), 256, 0, stream>>>(PV, PI, FAR);

  // --- 4 GIN layers ---
  const float* h = x;
  for (int c = 0; c < 4; ++c) {
    int Cdim = (c == 0) ? N_FEAT : NHID;
    size_t hc = (size_t)N_NODES * Cdim;
    fill_f32<<<igrid(Cdim, 256), 256, 0, stream>>>(COLSUM, (size_t)Cdim);
    fill_f32<<<igrid(hc, 256), 256, 0, stream>>>(EXCL, hc);
    colsum_kernel<<<N_NODES / 64, 256, 0, stream>>>(h, COLSUM, Cdim);
    scatter_kernel<<<igrid(hc, 256), 256, 0, stream>>>(h, FAR, EXCL, Cdim);
    agg_kernel<<<igrid(hc, 256), 256, 0, stream>>>(h, COLSUM, EXCL, ZH, Cdim);
    // t = relu(z @ w1 + b1) -> f16
    gemm_kernel<false, true><<<(N_NODES / 16) * (NHID / 16) / 8, 256, 0, stream>>>(
        ZH, W1T[c], nullptr, nullptr, conv_b1[c], 1.0f, nullptr, TH, N_NODES, NHID, Cdim);
    // x_c = relu(t @ w2 + b2) -> f32
    gemm_kernel<false, true><<<(N_NODES / 16) * (NHID / 16) / 8, 256, 0, stream>>>(
        TH, W2T[c], nullptr, nullptr, conv_b2[c], 1.0f, X[c], nullptr, N_NODES, NHID, NHID);
    h = X[c];
  }

  // --- residuals + stack stats ---
  stats_kernel<<<igrid(NH, 256), 256, 0, stream>>>(X[0], X[1], X[2], X[3],
                                                   S[0], S[1], S[2], S[3], S[4], PARTIAL);

  // --- 2-layer LSTM over the 5-step stack ---
  fill_f32<<<igrid(NH, 256), 256, 0, stream>>>(C0, NH);
  fill_f32<<<igrid(NH, 256), 256, 0, stream>>>(C1, NH);
  fill_f32<<<igrid(NH, 256), 256, 0, stream>>>(LSUM, NH);
  fill_f16<<<igrid(NH, 256), 256, 0, stream>>>(H0, NH);
  fill_f16<<<igrid(NH, 256), 256, 0, stream>>>(H1, NH);
  const int lstm_blocks = (N_NODES / 16) * (1024 / 16) / 8;
  for (int t = 0; t < 5; ++t) {
    // layer 0: G = s_t @ wih0^T + h0 @ whh0^T + (bih0+bhh0)
    gemm_kernel<true, false><<<lstm_blocks, 256, 0, stream>>>(
        S[t], WIH[0], H0, WHH[0], BSUM[0], 1.0f, G, nullptr, N_NODES, 1024, NHID);
    lstm_gate_kernel<<<igrid(NH, 256), 256, 0, stream>>>(G, C0, H0, nullptr);
    // layer 1: consumes layer-0 output of this step
    gemm_kernel<true, false><<<lstm_blocks, 256, 0, stream>>>(
        H0, WIH[1], H1, WHH[1], BSUM[1], 1.0f, G, nullptr, N_NODES, 1024, NHID);
    lstm_gate_kernel<<<igrid(NH, 256), 256, 0, stream>>>(G, C1, H1, LSUM);
  }

  // --- final: out = relu((avg + max + lstm_mean) @ mlp_w + 3*mlp_b) ---
  final_s_kernel<<<igrid(NH, 256), 256, 0, stream>>>(PARTIAL, LSUM, SH);
  gemm_kernel<false, true><<<(N_NODES / 16) * (NHID / 16) / 8, 256, 0, stream>>>(
      SH, MLPT, nullptr, nullptr, mlp_b, 3.0f, (float*)d_out, nullptr, N_NODES, NHID, NHID);
}
